// GAT_84310208020884
// MI455X (gfx1250) — matrix operand order repaired and checked by reading, not compiled, not run on hardware
//
#include <hip/hip_runtime.h>
#include <math.h>

// ---------------------------------------------------------------------------
// GAT (2-layer, PyG GATConv semantics) for MI455X / gfx1250.
//   GEMMs     : V_WMMA_F32_16X16X32_BF16; each wave computes a 16 x (NT*16)
//               strip, sharing one A fragment across NT accumulators
//               (4x less A traffic, ~2x higher wmma-per-vmem density).
//   Edge ops  : 3-pass segment softmax + weighted scatter with f32 atomics
//               (scatter targets fit in the 192MB L2, atomics resolve there).
// ---------------------------------------------------------------------------

typedef __attribute__((ext_vector_type(16))) __bf16 v16bf;
typedef __attribute__((ext_vector_type(8)))  __bf16 v8bf;
typedef __attribute__((ext_vector_type(8)))  float  v8f;

#define NEG_SLOPE 0.2f

__device__ __forceinline__ float lrelu_f(float v) { return v > 0.f ? v : v * NEG_SLOPE; }
__device__ __forceinline__ float elu_f(float v)   { return v > 0.f ? v : (expf(v) - 1.f); }

__device__ __forceinline__ void atomicMaxF32(float* addr, float val) {
  int* ai = (int*)addr;
  int old = __float_as_int(*addr);
  while (__int_as_float(old) < val) {
    int assumed = old;
    old = atomicCAS(ai, assumed, __float_as_int(val));
    if (old == assumed) break;
  }
}

// ----------------------------- elementwise ---------------------------------

__global__ void k_cvt_bf16(const float* __restrict__ in, __bf16* __restrict__ out, long n) {
  long i = (long)blockIdx.x * blockDim.x + threadIdx.x;
  if (i < n) out[i] = (__bf16)in[i];
}

// Bt[n*K + k] = (bf16)B[k*N + n]   (row-major [K,N] -> column-major bf16)
__global__ void k_transpose_bf16(const float* __restrict__ B, __bf16* __restrict__ Bt,
                                 int K, int N) {
  int i = blockIdx.x * blockDim.x + threadIdx.x;
  if (i >= K * N) return;
  int k = i / N, n = i % N;
  Bt[(long)n * K + k] = (__bf16)B[i];
}

__global__ void k_fill_f32(float* __restrict__ p, float v, long n) {
  long i = (long)blockIdx.x * blockDim.x + threadIdx.x;
  if (i < n) p[i] = v;
}

// out_bf[i] = (bf16) elu(in[i] + bias[i % F])
__global__ void k_bias_elu_bf16(const float* __restrict__ in, const float* __restrict__ bias,
                                __bf16* __restrict__ out, long n, int F) {
  long i = (long)blockIdx.x * blockDim.x + threadIdx.x;
  if (i >= n) return;
  out[i] = (__bf16)elu_f(in[i] + bias[(int)(i % F)]);
}

// ------------------------------ WMMA GEMM ----------------------------------
// C[M,N] (f32) = A[M,K] (bf16, row-major) x Bt (bf16, column-major: Bt[n*K+k])
// One wave computes a 16 x (NT*16) strip: one A fragment feeds NT WMMAs per
// K-step. Requires M % 16 == 0, N % (16*NT) == 0, K % 32 == 0.
template <int NT>
__global__ void k_wmma_gemm(const __bf16* __restrict__ A, const __bf16* __restrict__ Bt,
                            float* __restrict__ C, const float* __restrict__ bias,
                            int M, int N, int K) {
  int wave = (int)((blockIdx.x * blockDim.x + threadIdx.x) >> 5);
  int lane = threadIdx.x & 31;
  int groupsN = (N >> 4) / NT;
  int total   = (M >> 4) * groupsN;
  if (wave >= total) return;
  int tm = wave / groupsN;
  int tg = wave % groupsN;
  int half = lane >> 4;     // 0: lanes 0-15, 1: lanes 16-31
  int l16  = lane & 15;

  // A fragment: lane holds row (tm*16 + l16); per K-step, two contiguous
  // 8-element chunks at K offsets {half*8, 16+half*8}  (ISA 7.12.2).
  const __bf16* pa = A + (long)(tm * 16 + l16) * K;
  // B fragments: lane holds column ((tg*NT + t)*16 + l16); per K-step, two
  // 8-element chunks at K offsets {half*8, 16+half*8} (column-major storage).
  const __bf16* pb = Bt + (long)(tg * NT * 16 + l16) * K;

  v8f acc[NT];
#pragma unroll
  for (int t = 0; t < NT; ++t) acc[t] = (v8f){};

  for (int ko = 0; ko < K; ko += 32) {
    v8bf a0 = *(const v8bf*)(pa + ko + half * 8);
    v8bf a1 = *(const v8bf*)(pa + ko + 16 + half * 8);
    v16bf a;
#pragma unroll
    for (int i = 0; i < 8; ++i) { a[i] = a0[i]; a[i + 8] = a1[i]; }
#pragma unroll
    for (int t = 0; t < NT; ++t) {
      v8bf b0 = *(const v8bf*)(pb + (long)t * 16 * K + ko + half * 8);
      v8bf b1 = *(const v8bf*)(pb + (long)t * 16 * K + ko + 16 + half * 8);
      v16bf b;
#pragma unroll
      for (int i = 0; i < 8; ++i) { b[i] = b0[i]; b[i + 8] = b1[i]; }
      acc[t] = __builtin_amdgcn_wmma_f32_16x16x32_bf16(
          /*neg_a=*/false, a, /*neg_b=*/false, b,
          /*c_mod=*/(short)0, acc[t], /*reuse_a=*/false, /*reuse_b=*/false);
    }
  }

  // D layout: VGPR i -> row (i + 8*half), col l16.
#pragma unroll
  for (int t = 0; t < NT; ++t) {
    int col = (tg * NT + t) * 16 + l16;
    float bv = bias ? bias[col] : 0.f;
    float* pc = C + (long)(tm * 16 + half * 8) * N + col;
#pragma unroll
    for (int i = 0; i < 8; ++i) pc[(long)i * N] = acc[t][i] + bv;
  }
}

// ------------------------- attention coefficients --------------------------
// as_[n,h] = sum_c h[n,h,c]*att_src[h,c];  ad_ likewise.
__global__ void k_att(const float* __restrict__ h, const float* __restrict__ ws,
                      const float* __restrict__ wd, float* __restrict__ as_,
                      float* __restrict__ ad_, int Nn, int H, int C) {
  int i = blockIdx.x * blockDim.x + threadIdx.x;
  if (i >= Nn * H) return;
  int n = i / H, hh = i % H;
  const float* hp = h + (long)n * H * C + hh * C;
  const float* sp = ws + hh * C;
  const float* dp = wd + hh * C;
  float s = 0.f, d = 0.f;
  for (int c = 0; c < C; ++c) { s += hp[c] * sp[c]; d += hp[c] * dp[c]; }
  as_[i] = s;
  ad_[i] = d;
}

// ------------------------------ edge passes --------------------------------
// edge e < E: src = ei[e], dst = ei[E+e]; e >= E: self-loop (src=dst=e-E).

__global__ void k_edge_logits(const long long* __restrict__ ei, int E, int Etot, int H,
                              const float* __restrict__ as_, const float* __restrict__ ad_,
                              float* __restrict__ ebuf, float* __restrict__ m) {
  int idx = blockIdx.x * blockDim.x + threadIdx.x;
  if (idx >= Etot * H) return;
  int e = idx / H, hh = idx % H;
  long long s, d;
  if (e < E) { s = ei[e]; d = ei[E + e]; } else { s = d = e - E; }
  float v = lrelu_f(as_[s * H + hh] + ad_[d * H + hh]);
  ebuf[idx] = v;
  atomicMaxF32(&m[d * H + hh], v);
}

__global__ void k_edge_exp(const long long* __restrict__ ei, int E, int Etot, int H,
                           const float* __restrict__ m, float* __restrict__ ebuf,
                           float* __restrict__ den) {
  int idx = blockIdx.x * blockDim.x + threadIdx.x;
  if (idx >= Etot * H) return;
  int e = idx / H, hh = idx % H;
  long long d;
  if (e < E) { d = ei[E + e]; } else { d = e - E; }
  float ex = expf(ebuf[idx] - m[d * H + hh]);
  ebuf[idx] = ex;
  atomicAdd(&den[d * H + hh], ex);
}

// out[dst, :] += h[src, :] * alpha ; one thread per (edge, 4-channel group).
__global__ void k_edge_aggr(const long long* __restrict__ ei, int E, int Etot, int H, int C,
                            const float* __restrict__ hfeat, const float* __restrict__ ebuf,
                            const float* __restrict__ den, float* __restrict__ out) {
  int HC = H * C;
  int groups = HC >> 2;
  long idx = (long)blockIdx.x * blockDim.x + threadIdx.x;
  if (idx >= (long)Etot * groups) return;
  int e = (int)(idx / groups);
  int g = (int)(idx % groups);
  long long s, d;
  if (e < E) { s = ei[e]; d = ei[E + e]; } else { s = d = e - E; }
  int ch = g * 4;
  int hh = ch / C;
  float alpha = ebuf[(long)e * H + hh] / den[d * H + hh];
  const float4 hv = *(const float4*)(hfeat + (long)s * HC + ch);
  float* op = out + (long)d * HC + ch;
  atomicAdd(op + 0, hv.x * alpha);
  atomicAdd(op + 1, hv.y * alpha);
  atomicAdd(op + 2, hv.z * alpha);
  atomicAdd(op + 3, hv.w * alpha);
}

// ---------------------------------------------------------------------------

extern "C" void kernel_launch(void* const* d_in, const int* in_sizes, int n_in,
                              void* d_out, int out_size, void* d_ws, size_t ws_size,
                              hipStream_t stream) {
  (void)in_sizes; (void)n_in; (void)out_size; (void)ws_size;

  const int Nn   = 50000;
  const int E    = 800000;
  const int Etot = E + Nn;     // self loops appended
  const int IN   = 128;
  const int H1   = 4, C1 = 64, HC1 = H1 * C1;  // 256
  const int C2   = 64;
  const int OUT  = 16;

  const float*     x    = (const float*)d_in[0];
  const long long* ei   = (const long long*)d_in[1];
  // d_in[2] = edge_attr, unused (edge_dim=None)
  const float* W1   = (const float*)d_in[3];
  const float* aS1  = (const float*)d_in[4];
  const float* aD1  = (const float*)d_in[5];
  const float* b1   = (const float*)d_in[6];
  const float* W2   = (const float*)d_in[7];
  const float* aS2  = (const float*)d_in[8];
  const float* aD2  = (const float*)d_in[9];
  const float* b2   = (const float*)d_in[10];
  const float* WL   = (const float*)d_in[11];
  const float* bL   = (const float*)d_in[12];

  // ---- workspace carve (256B aligned chunks) ----
  char* wp = (char*)d_ws;
  auto carve = [&](size_t bytes) -> void* {
    void* p = (void*)wp;
    wp += (bytes + 255) & ~(size_t)255;
    return p;
  };
  __bf16* x_bf   = (__bf16*)carve((size_t)Nn * IN * 2);
  __bf16* W1t    = (__bf16*)carve((size_t)IN * HC1 * 2);
  __bf16* W2t    = (__bf16*)carve((size_t)HC1 * C2 * 2);
  __bf16* WLt    = (__bf16*)carve((size_t)C2 * OUT * 2);
  float*  h1     = (float*)carve((size_t)Nn * HC1 * 4);
  float*  as1    = (float*)carve((size_t)Nn * H1 * 4);
  float*  ad1    = (float*)carve((size_t)Nn * H1 * 4);
  float*  m1     = (float*)carve((size_t)Nn * H1 * 4);
  float*  den1   = (float*)carve((size_t)Nn * H1 * 4);
  float*  ebuf   = (float*)carve((size_t)Etot * H1 * 4);   // reused for layer 2
  float*  out1   = (float*)carve((size_t)Nn * HC1 * 4);
  __bf16* h1e_bf = (__bf16*)carve((size_t)Nn * HC1 * 2);
  float*  h2     = (float*)carve((size_t)Nn * C2 * 4);
  float*  as2    = (float*)carve((size_t)Nn * 4);
  float*  ad2    = (float*)carve((size_t)Nn * 4);
  float*  m2     = (float*)carve((size_t)Nn * 4);
  float*  den2   = (float*)carve((size_t)Nn * 4);
  float*  out2   = (float*)carve((size_t)Nn * C2 * 4);
  __bf16* h2e_bf = (__bf16*)carve((size_t)Nn * C2 * 2);

  const int BLK = 256;
  auto nblk = [](long n, int b) { return (int)((n + b - 1) / b); };
  auto gemmB = [](int M, int N, int NT) {
    long waves = (long)(M / 16) * ((N / 16) / NT);
    return (int)((waves * 32 + 255) / 256);
  };

  // ---- stage inputs: x -> bf16, weights -> transposed bf16 ----
  k_cvt_bf16<<<nblk((long)Nn * IN, BLK), BLK, 0, stream>>>(x, x_bf, (long)Nn * IN);
  k_transpose_bf16<<<nblk((long)IN * HC1, BLK), BLK, 0, stream>>>(W1, W1t, IN, HC1);
  k_transpose_bf16<<<nblk((long)HC1 * C2, BLK), BLK, 0, stream>>>(W2, W2t, HC1, C2);
  k_transpose_bf16<<<nblk((long)C2 * OUT, BLK), BLK, 0, stream>>>(WL, WLt, C2, OUT);

  // ================= layer 1: GATConv(128 -> 4 x 64, concat) ===============
  k_wmma_gemm<4><<<gemmB(Nn, HC1, 4), BLK, 0, stream>>>(x_bf, W1t, h1, nullptr, Nn, HC1, IN);
  k_att<<<nblk((long)Nn * H1, BLK), BLK, 0, stream>>>(h1, aS1, aD1, as1, ad1, Nn, H1, C1);

  k_fill_f32<<<nblk((long)Nn * H1, BLK), BLK, 0, stream>>>(m1, -INFINITY, (long)Nn * H1);
  k_fill_f32<<<nblk((long)Nn * H1, BLK), BLK, 0, stream>>>(den1, 0.f, (long)Nn * H1);
  k_fill_f32<<<nblk((long)Nn * HC1, BLK), BLK, 0, stream>>>(out1, 0.f, (long)Nn * HC1);

  k_edge_logits<<<nblk((long)Etot * H1, BLK), BLK, 0, stream>>>(ei, E, Etot, H1, as1, ad1, ebuf, m1);
  k_edge_exp<<<nblk((long)Etot * H1, BLK), BLK, 0, stream>>>(ei, E, Etot, H1, m1, ebuf, den1);
  k_edge_aggr<<<nblk((long)Etot * (HC1 / 4), BLK), BLK, 0, stream>>>(ei, E, Etot, H1, C1,
                                                                    h1, ebuf, den1, out1);
  k_bias_elu_bf16<<<nblk((long)Nn * HC1, BLK), BLK, 0, stream>>>(out1, b1, h1e_bf,
                                                                 (long)Nn * HC1, HC1);

  // ================= layer 2: GATConv(256 -> 1 x 64) =======================
  k_wmma_gemm<4><<<gemmB(Nn, C2, 4), BLK, 0, stream>>>(h1e_bf, W2t, h2, nullptr, Nn, C2, HC1);
  k_att<<<nblk((long)Nn, BLK), BLK, 0, stream>>>(h2, aS2, aD2, as2, ad2, Nn, 1, C2);

  k_fill_f32<<<nblk((long)Nn, BLK), BLK, 0, stream>>>(m2, -INFINITY, (long)Nn);
  k_fill_f32<<<nblk((long)Nn, BLK), BLK, 0, stream>>>(den2, 0.f, (long)Nn);
  k_fill_f32<<<nblk((long)Nn * C2, BLK), BLK, 0, stream>>>(out2, 0.f, (long)Nn * C2);

  k_edge_logits<<<nblk((long)Etot, BLK), BLK, 0, stream>>>(ei, E, Etot, 1, as2, ad2, ebuf, m2);
  k_edge_exp<<<nblk((long)Etot, BLK), BLK, 0, stream>>>(ei, E, Etot, 1, m2, ebuf, den2);
  k_edge_aggr<<<nblk((long)Etot * (C2 / 4), BLK), BLK, 0, stream>>>(ei, E, Etot, 1, C2,
                                                                   h2, ebuf, den2, out2);
  k_bias_elu_bf16<<<nblk((long)Nn * C2, BLK), BLK, 0, stream>>>(out2, b2, h2e_bf,
                                                                (long)Nn * C2, C2);

  // ================= linear head: [N,64] @ [64,16] + bias ==================
  k_wmma_gemm<1><<<gemmB(Nn, OUT, 1), BLK, 0, stream>>>(h2e_bf, WLt, (float*)d_out, bL,
                                                        Nn, OUT, C2);
}
